// PatternMoE_85083302133970
// MI455X (gfx1250) — matrix operand order repaired and checked by reading, not compiled, not run on hardware
//
#include <hip/hip_runtime.h>
#include <math.h>

// ---------------- constants (match reference) ----------------
constexpr int Hc   = 512;   // hidden
constexpr int Ec   = 8;     // experts
constexpr int TOPK = 4;
constexpr int NHc  = 4;     // heads
constexpr int HDc  = 128;   // head dim
constexpr int Bc   = 4;     // batch
constexpr int LVL0 = 2048, LVL1 = 1024, LVL2 = 512;

typedef float  v8f   __attribute__((ext_vector_type(8)));
typedef __bf16 v16bf __attribute__((ext_vector_type(16)));

__device__ __forceinline__ unsigned short f2bf(float f) {
  unsigned int u = __float_as_uint(f);
  u += 0x7FFFu + ((u >> 16) & 1u);          // round-to-nearest-even
  return (unsigned short)(u >> 16);
}
__device__ __forceinline__ float gelu_exact(float x) {
  return 0.5f * x * (1.0f + erff(x * 0.7071067811865475f));
}

// ---------------- WMMA GEMM ----------------
// C[M,N] = epilogue( scale * (A[M,K] @ B[K,N]) + bias ) (+ residual)
// A: f32 row-major, row stride lda.  B: f32, element (k,n) at B + k*ldbk + n*ldbn.
// Block tile 128x64, K step 64. 8 waves in 4x2 grid; each wave owns a 2x2 grid
// of 16x16 WMMA tiles -> 8 v_wmma_f32_16x16x32_bf16 per staging round.
#define TM 128
#define TN 64
#define TK 64
#define LDP 72   // padded LDS row stride in halfs (multiple of 8 -> 16B aligned)

template<bool GELU>
__global__ __launch_bounds__(256)
void gemm_bf16_wmma(const float* __restrict__ A, int lda,
                    const float* __restrict__ Bm, long long ldbk, long long ldbn,
                    const float* __restrict__ bias,
                    const float* __restrict__ res, int ldres,
                    float* __restrict__ C, int ldc,
                    int M, int N, int Kd, float scale)
{
  __shared__ __align__(16) unsigned short As[TM * LDP];   // [row][k]
  __shared__ __align__(16) unsigned short Bs[TN * LDP];   // [n][k] (transposed)

  const int tid  = threadIdx.x;
  const int m0   = blockIdx.y * TM;
  const int n0   = blockIdx.x * TN;
  const int wave = tid >> 5, lane = tid & 31;
  const int wm = wave >> 1, wn = wave & 1;        // 4x2 wave grid (32x32 per wave)
  const int g  = lane >> 4, ml = lane & 15;

  v8f acc00 = {}, acc01 = {}, acc10 = {}, acc11 = {};

  for (int k0 = 0; k0 < Kd; k0 += TK) {
    // ---- stage A tile: 128 rows x 64 halfs = 1024 chunks of 8, 4 per thread ----
    #pragma unroll
    for (int i = 0; i < 4; ++i) {
      const int c   = i * 256 + tid;
      const int row = c >> 3;
      const int kch = (c & 7) * 8;
      union { unsigned short u[8]; uint4 q; } t;
      const int gm = m0 + row;
      if (gm < M) {
        const float* src = A + (long long)gm * lda + k0 + kch;
        #pragma unroll
        for (int e = 0; e < 8; ++e) t.u[e] = f2bf(src[e]);
        if (i == 0 && k0 + TK < Kd) __builtin_prefetch(src + TK, 0, 3); // global_prefetch_b8
      } else {
        #pragma unroll
        for (int e = 0; e < 8; ++e) t.u[e] = 0;
      }
      *(uint4*)&As[row * LDP + kch] = t.q;
    }
    // ---- stage B tile transposed: 64 n x 64 k = 512 chunks of 8, 2 per thread ----
    #pragma unroll
    for (int i = 0; i < 2; ++i) {
      const int c   = i * 256 + tid;
      const int n   = c >> 3;
      const int kch = (c & 7) * 8;
      union { unsigned short u[8]; uint4 q; } t;
      const int gn = n0 + n;
      if (gn < N) {
        #pragma unroll
        for (int e = 0; e < 8; ++e)
          t.u[e] = f2bf(Bm[(long long)(k0 + kch + e) * ldbk + (long long)gn * ldbn]);
      } else {
        #pragma unroll
        for (int e = 0; e < 8; ++e) t.u[e] = 0;
      }
      *(uint4*)&Bs[n * LDP + kch] = t.q;
    }
    __syncthreads();

    // ---- 2 K sub-steps x 2x2 tiles = 8 WMMA per round ----
    #pragma unroll
    for (int kk = 0; kk < TK; kk += 32) {
      union { v16bf v; uint4 q[2]; } fa0, fa1, fb0, fb1;
      const int a0 = (wm * 32 + ml) * LDP + kk;        // A tile row m-block 0
      const int a1 = (wm * 32 + 16 + ml) * LDP + kk;   // A tile row m-block 1
      fa0.q[0] = *(const uint4*)&As[a0 + 16 * g];
      fa0.q[1] = *(const uint4*)&As[a0 + 16 * g + 8];
      fa1.q[0] = *(const uint4*)&As[a1 + 16 * g];
      fa1.q[1] = *(const uint4*)&As[a1 + 16 * g + 8];
      const int b0 = (wn * 32 + ml) * LDP + kk;        // B tile n-block 0
      const int b1 = (wn * 32 + 16 + ml) * LDP + kk;   // B tile n-block 1
      fb0.q[0] = *(const uint4*)&Bs[b0 + 16 * g];
      fb0.q[1] = *(const uint4*)&Bs[b0 + 16 * g + 8];
      fb1.q[0] = *(const uint4*)&Bs[b1 + 16 * g];
      fb1.q[1] = *(const uint4*)&Bs[b1 + 16 * g + 8];

      acc00 = __builtin_amdgcn_wmma_f32_16x16x32_bf16(false, fa0.v, false, fb0.v, (short)0, acc00, false, false);
      acc01 = __builtin_amdgcn_wmma_f32_16x16x32_bf16(false, fa0.v, false, fb1.v, (short)0, acc01, false, false);
      acc10 = __builtin_amdgcn_wmma_f32_16x16x32_bf16(false, fa1.v, false, fb0.v, (short)0, acc10, false, false);
      acc11 = __builtin_amdgcn_wmma_f32_16x16x32_bf16(false, fa1.v, false, fb1.v, (short)0, acc11, false, false);
    }
    __syncthreads();
  }

  // ---- epilogue: scale, bias, GELU, residual ----
  v8f* accs[4] = { &acc00, &acc01, &acc10, &acc11 };
  #pragma unroll
  for (int tmb = 0; tmb < 2; ++tmb) {
    #pragma unroll
    for (int tnb = 0; tnb < 2; ++tnb) {
      const v8f& a = *accs[tmb * 2 + tnb];
      const int cn = n0 + wn * 32 + tnb * 16 + ml;
      if (cn < N) {
        const float bv = bias ? bias[cn] : 0.0f;
        #pragma unroll
        for (int r = 0; r < 8; ++r) {
          const int cm = m0 + wm * 32 + tmb * 16 + 8 * g + r;
          if (cm < M) {
            float x = a[r] * scale + bv;
            if (GELU) x = gelu_exact(x);
            if (res) x += res[(long long)cm * ldres + cn];
            C[(long long)cm * ldc + cn] = x;
          }
        }
      }
    }
  }
}

// ---------------- elementwise / reduction kernels ----------------
__global__ __launch_bounds__(256)
void layernorm_rows(const float* __restrict__ in, float* __restrict__ out,
                    const float* __restrict__ gamma, const float* __restrict__ beta)
{
  const long long row = blockIdx.x;
  const float* x = in + row * Hc;
  float* y = out + row * Hc;
  __shared__ float red[256];
  float s = 0.f;
  for (int i = threadIdx.x; i < Hc; i += 256) s += x[i];
  red[threadIdx.x] = s; __syncthreads();
  for (int t = 128; t > 0; t >>= 1) { if (threadIdx.x < t) red[threadIdx.x] += red[threadIdx.x + t]; __syncthreads(); }
  const float mean = red[0] * (1.0f / Hc);
  __syncthreads();
  float v = 0.f;
  for (int i = threadIdx.x; i < Hc; i += 256) { float d = x[i] - mean; v += d * d; }
  red[threadIdx.x] = v; __syncthreads();
  for (int t = 128; t > 0; t >>= 1) { if (threadIdx.x < t) red[threadIdx.x] += red[threadIdx.x + t]; __syncthreads(); }
  const float inv = rsqrtf(red[0] * (1.0f / Hc) + 1e-5f);
  for (int i = threadIdx.x; i < Hc; i += 256)
    y[i] = (x[i] - mean) * inv * gamma[i] + beta[i];
}

__global__ __launch_bounds__(256)
void softmax_rows(float* __restrict__ data, int ncols)
{
  const long long row = blockIdx.x;
  float* x = data + row * (long long)ncols;
  __shared__ float red[256];
  float mx = -3.4e38f;
  for (int i = threadIdx.x; i < ncols; i += 256) mx = fmaxf(mx, x[i]);
  red[threadIdx.x] = mx; __syncthreads();
  for (int t = 128; t > 0; t >>= 1) { if (threadIdx.x < t) red[threadIdx.x] = fmaxf(red[threadIdx.x], red[threadIdx.x + t]); __syncthreads(); }
  mx = red[0]; __syncthreads();
  float s = 0.f;
  for (int i = threadIdx.x; i < ncols; i += 256) { float e = __expf(x[i] - mx); x[i] = e; s += e; }
  red[threadIdx.x] = s; __syncthreads();
  for (int t = 128; t > 0; t >>= 1) { if (threadIdx.x < t) red[threadIdx.x] += red[threadIdx.x + t]; __syncthreads(); }
  const float inv = 1.0f / red[0];
  for (int i = threadIdx.x; i < ncols; i += 256) x[i] *= inv;
}

__global__ __launch_bounds__(256)
void transpose_hl(const float* __restrict__ x, float* __restrict__ xt, int L, long long total)
{ // x (B,H,L) -> xt (B,L,H)
  long long i = (long long)blockIdx.x * 256 + threadIdx.x;
  if (i >= total) return;
  int h = (int)(i % Hc);
  long long t = i / Hc;
  int l = (int)(t % L);
  int b = (int)(t / L);
  xt[i] = x[((long long)b * Hc + h) * L + l];
}

__global__ __launch_bounds__(256)
void final_out(const float* __restrict__ x, const float* __restrict__ acc,
               float* __restrict__ y, int L, long long total)
{ // y (B,H,L) = x (B,H,L) + acc (B,L,H)
  long long i = (long long)blockIdx.x * 256 + threadIdx.x;
  if (i >= total) return;
  int l = (int)(i % L);
  long long t = i / L;
  int h = (int)(t % Hc);
  int b = (int)(t / Hc);
  y[i] = x[i] + acc[((long long)b * L + l) * Hc + h];
}

__global__ __launch_bounds__(256)
void mean_lastdim(const float* __restrict__ x, float* __restrict__ out, int L)
{
  int i = blockIdx.x * 256 + threadIdx.x;
  if (i >= Bc * Hc) return;
  const float* p = x + (long long)i * L;
  float s = 0.f;
  for (int l = 0; l < L; ++l) s += p[l];
  out[i] = s / (float)L;
}

__global__ __launch_bounds__(256)
void rope_kernel(float* __restrict__ q, int L, long long total)
{ // q (B*L, H) viewed as (B,L,NH,HD); total = B*L*NH*64
  long long idx = (long long)blockIdx.x * 256 + threadIdx.x;
  if (idx >= total) return;
  const int j = (int)(idx & 63);
  long long t = idx >> 6;
  const int h = (int)(t & (NHc - 1));
  const long long row = t >> 2;
  const int l = (int)(row % L);
  const float fr = powf(10000.0f, -(float)j / 64.0f);
  const float ang = (float)l * fr;
  const float c = cosf(ang), s = sinf(ang);
  float* base = q + row * Hc + h * HDc;
  const float x1 = base[j], x2 = base[j + 64];
  base[j]      = x1 * c - x2 * s;
  base[j + 64] = x1 * s + x2 * c;
}

__global__ __launch_bounds__(256)
void build_patch(const float* __restrict__ x, float* __restrict__ patch, int L, int d, long long total)
{ // patch[b, l, t*H+c] = x[b, c, l+(t-1)*d] (zero padded)
  long long i = (long long)blockIdx.x * 256 + threadIdx.x;
  if (i >= total) return;
  const int c3 = (int)(i % (3 * Hc));
  long long t2 = i / (3 * Hc);
  const int l = (int)(t2 % L);
  const int b = (int)(t2 / L);
  const int t = c3 / Hc, c = c3 % Hc;
  const int src = l + (t - 1) * d;
  patch[i] = (src >= 0 && src < L) ? x[((long long)b * Hc + c) * L + src] : 0.0f;
}

__global__ __launch_bounds__(256)
void pack_w1(const float* __restrict__ w1, float* __restrict__ p1)
{ // w1 (2H,H,3) -> p1[(t*H+c)][o] row-major N=2H
  long long i = (long long)blockIdx.x * 256 + threadIdx.x;
  const long long total = 3LL * Hc * 2 * Hc;
  if (i >= total) return;
  const int o = (int)(i % (2 * Hc));
  const int k = (int)(i / (2 * Hc));
  const int t = k / Hc, c = k % Hc;
  p1[i] = w1[((long long)o * Hc + c) * 3 + t];
}

__global__ __launch_bounds__(256)
void pack_w2(const float* __restrict__ w2, float* __restrict__ p2)
{ // w2 (H,2H,1) -> p2[c][o] row-major N=H
  long long i = (long long)blockIdx.x * 256 + threadIdx.x;
  const long long total = 2LL * Hc * Hc;
  if (i >= total) return;
  const int o = (int)(i % Hc);
  const int c = (int)(i / Hc);
  p2[i] = w2[(long long)o * 2 * Hc + c];
}

__global__ __launch_bounds__(256)
void axpy_expert(float* __restrict__ acc, const float* __restrict__ eo,
                 const float* __restrict__ w, int e, long long LH, long long total)
{
  long long i = (long long)blockIdx.x * 256 + threadIdx.x;
  if (i >= total) return;
  const int b = (int)(i / LH);
  acc[i] += w[b * Ec + e] * eo[i];
}

__global__ __launch_bounds__(256)
void fill_f32(float* __restrict__ p, float v, long long n)
{
  long long i = (long long)blockIdx.x * 256 + threadIdx.x;
  if (i < n) p[i] = v;
}

__global__ void router_finalize(const float* __restrict__ logits,
                                float* __restrict__ probs_out,
                                float* __restrict__ w,
                                float* __restrict__ counts)
{
  const int b = threadIdx.x;
  if (b >= Bc) return;
  float p[Ec];
  float mx = -3.4e38f;
  for (int e = 0; e < Ec; ++e) { p[e] = logits[b * Ec + e]; mx = fmaxf(mx, p[e]); }
  float s = 0.f;
  for (int e = 0; e < Ec; ++e) { p[e] = __expf(p[e] - mx); s += p[e]; }
  const float inv = 1.0f / s;
  for (int e = 0; e < Ec; ++e) { p[e] *= inv; probs_out[b * Ec + e] = p[e]; }
  int  sel[TOPK]; float tp[TOPK]; bool used[Ec] = {};
  float tsum = 0.f;
  for (int k = 0; k < TOPK; ++k) {
    int best = 0; float bv = -1.f;
    for (int e = 0; e < Ec; ++e) if (!used[e] && p[e] > bv) { bv = p[e]; best = e; }
    used[best] = true; sel[k] = best; tp[k] = bv; tsum += bv;
  }
  for (int e = 0; e < Ec; ++e) w[b * Ec + e] = 0.f;
  for (int k = 0; k < TOPK; ++k) {
    w[b * Ec + sel[k]] = tp[k] / tsum;
    atomicAdd(&counts[sel[k]], 1.0f);
  }
}

__global__ void usage_kernel(const float* __restrict__ counts, float* __restrict__ usage)
{
  const int e = threadIdx.x;
  if (e < Ec) usage[e] = counts[e] / 12.0f;   // 3 levels * B=4
}

// ---------------- host orchestration ----------------
extern "C" void kernel_launch(void* const* d_in, const int* in_sizes, int n_in,
                              void* d_out, int out_size, void* d_ws, size_t ws_size,
                              hipStream_t stream)
{
  (void)in_sizes; (void)n_in; (void)out_size; (void)ws_size;

  // input order: x0,x1,x2 then params leaves in declaration order
  const float* X[3] = { (const float*)d_in[0], (const float*)d_in[1], (const float*)d_in[2] };
  const float* norm_g = (const float*)d_in[3];
  const float* norm_b = (const float*)d_in[4];
  const float* rW1    = (const float*)d_in[5];
  const float* rb1    = (const float*)d_in[6];
  const float* rW2    = (const float*)d_in[7];
  const float* rb2    = (const float*)d_in[8];
  auto lw = [&](int i, int j) { return (const float*)d_in[9 + i * 4 + j]; };   // w1,b1,w2,b2
  auto gw = [&](int j, int k) { return (const float*)d_in[25 + j * 14 + k]; }; // ln_g..c2

  // workspace arena (f32 elements), sized for Lmax=2048
  float* wsf = (float*)d_ws;
  const long long BLmax = (long long)Bc * LVL0;       // 8192
  long long off = 0;
  auto alloc = [&](long long n) { float* p = wsf + off; off += n; return p; };
  float* xt     = alloc(BLmax * Hc);
  float* acc    = alloc(BLmax * Hc);
  float* hG     = alloc(BLmax * Hc);
  float* qb     = alloc(BLmax * Hc);      // reused as h2 after attention
  float* kb     = alloc(BLmax * Hc);
  float* vb     = alloc(BLmax * Hc);
  float* ob     = alloc(BLmax * Hc);
  float* scores = alloc((long long)LVL0 * LVL0);
  float* mid    = alloc(BLmax * 2 * Hc);
  float* eo     = alloc(BLmax * Hc);
  float* patch  = alloc(BLmax * 3 * Hc);
  float* p1base = alloc(4LL * 3 * Hc * 2 * Hc);
  float* p2base = alloc(4LL * 2 * Hc * Hc);
  float* rmean  = alloc(Bc * Hc);
  float* rln    = alloc(Bc * Hc);
  float* rmid   = alloc(Bc * Hc);
  float* logits = alloc(Bc * Ec);
  float* wbuf   = alloc(Bc * Ec);
  float* counts = alloc(Ec);

  // output layout: ys0, ys1, ys2, probs0..2, usage
  float* out = (float*)d_out;
  float* ys[3] = { out, out + 4LL * Hc * LVL0, out + 4LL * Hc * (LVL0 + LVL1) };
  float* probsOut = out + 4LL * Hc * (LVL0 + LVL1 + LVL2);
  float* usageOut = probsOut + 3 * Bc * Ec;

  auto blocks = [](long long n) { return (unsigned)((n + 255) / 256); };
  auto gemm = [&](const float* A, int lda, const float* B, long long ldbk, long long ldbn,
                  const float* bias, const float* res, int ldres,
                  float* C, int ldc, int M, int N, int Kd, float scale, bool gelu) {
    dim3 grid((N + TN - 1) / TN, (M + TM - 1) / TM);
    if (gelu)
      gemm_bf16_wmma<true><<<grid, 256, 0, stream>>>(A, lda, B, ldbk, ldbn, bias, res, ldres, C, ldc, M, N, Kd, scale);
    else
      gemm_bf16_wmma<false><<<grid, 256, 0, stream>>>(A, lda, B, ldbk, ldbn, bias, res, ldres, C, ldc, M, N, Kd, scale);
  };

  // ---- once per launch: pack local-expert weights, zero counts ----
  for (int i = 0; i < 4; ++i) {
    pack_w1<<<blocks(3LL * Hc * 2 * Hc), 256, 0, stream>>>(lw(i, 0), p1base + (long long)i * 3 * Hc * 2 * Hc);
    pack_w2<<<blocks(2LL * Hc * Hc), 256, 0, stream>>>(lw(i, 2), p2base + (long long)i * 2 * Hc * Hc);
  }
  fill_f32<<<1, 256, 0, stream>>>(counts, 0.0f, Ec);

  const int Ls[3] = { LVL0, LVL1, LVL2 };
  const int dils[4] = { 2, 4, 8, 2 };
  const float iscale = 0.08838834764831845f;  // 1/sqrt(128)

  for (int lvl = 0; lvl < 3; ++lvl) {
    const int L = Ls[lvl];
    const long long BL = (long long)Bc * L;
    const float* x = X[lvl];

    transpose_hl<<<blocks(BL * Hc), 256, 0, stream>>>(x, xt, L, BL * Hc);
    fill_f32<<<blocks(BL * Hc), 256, 0, stream>>>(acc, 0.0f, BL * Hc);

    // ---- router ----
    mean_lastdim<<<blocks(Bc * Hc), 256, 0, stream>>>(x, rmean, L);
    layernorm_rows<<<Bc, 256, 0, stream>>>(rmean, rln, norm_g, norm_b);
    gemm(rln, Hc, rW1, Hc, 1, rb1, nullptr, 0, rmid, Hc, Bc, Hc, Hc, 1.0f, true);
    gemm(rmid, Hc, rW2, Ec, 1, rb2, nullptr, 0, logits, Ec, Bc, Ec, Hc, 1.0f, false);
    router_finalize<<<1, 32, 0, stream>>>(logits, probsOut + lvl * Bc * Ec, wbuf, counts);

    // ---- local experts (im2col + two WMMA GEMMs) ----
    for (int i = 0; i < 4; ++i) {
      build_patch<<<blocks(BL * 3 * Hc), 256, 0, stream>>>(x, patch, L, dils[i], BL * 3 * Hc);
      gemm(patch, 3 * Hc, p1base + (long long)i * 3 * Hc * 2 * Hc, 2 * Hc, 1,
           lw(i, 1), nullptr, 0, mid, 2 * Hc, (int)BL, 2 * Hc, 3 * Hc, 1.0f, true);
      gemm(mid, 2 * Hc, p2base + (long long)i * 2 * Hc * Hc, Hc, 1,
           lw(i, 3), nullptr, 0, eo, Hc, (int)BL, Hc, 2 * Hc, 1.0f, false);
      axpy_expert<<<blocks(BL * Hc), 256, 0, stream>>>(acc, eo, wbuf, i, (long long)L * Hc, BL * Hc);
    }

    // ---- global experts (LN + RoPE attention + FFN) ----
    for (int j = 0; j < 4; ++j) {
      const float *ln_g = gw(j, 0), *ln_b = gw(j, 1);
      const float *Wq = gw(j, 2), *bq = gw(j, 3), *Wk = gw(j, 4), *bk = gw(j, 5);
      const float *Wv = gw(j, 6), *bv = gw(j, 7), *Wo = gw(j, 8), *bo = gw(j, 9);
      const float *W1 = gw(j, 10), *c1 = gw(j, 11), *W2 = gw(j, 12), *c2 = gw(j, 13);

      layernorm_rows<<<(unsigned)BL, 256, 0, stream>>>(xt, hG, ln_g, ln_b);
      gemm(hG, Hc, Wq, Hc, 1, bq, nullptr, 0, qb, Hc, (int)BL, Hc, Hc, 1.0f, false);
      gemm(hG, Hc, Wk, Hc, 1, bk, nullptr, 0, kb, Hc, (int)BL, Hc, Hc, 1.0f, false);
      gemm(hG, Hc, Wv, Hc, 1, bv, nullptr, 0, vb, Hc, (int)BL, Hc, Hc, 1.0f, false);
      rope_kernel<<<blocks(BL * NHc * 64), 256, 0, stream>>>(qb, L, BL * NHc * 64);
      rope_kernel<<<blocks(BL * NHc * 64), 256, 0, stream>>>(kb, L, BL * NHc * 64);

      for (int b = 0; b < Bc; ++b) {
        for (int hh = 0; hh < NHc; ++hh) {
          const long long boff = (long long)b * L * Hc + hh * HDc;
          // scores = (q_h @ k_h^T) / sqrt(hd)
          gemm(qb + boff, Hc, kb + boff, 1, Hc, nullptr, nullptr, 0,
               scores, L, L, L, HDc, iscale, false);
          softmax_rows<<<(unsigned)L, 256, 0, stream>>>(scores, L);
          // o_h = attn @ v_h
          gemm(scores, L, vb + boff, Hc, 1, nullptr, nullptr, 0,
               ob + boff, Hc, L, HDc, L, 1.0f, false);
        }
      }
      // h2 = hG + (o @ Wo + bo)  (reuse qb as h2)
      gemm(ob, Hc, Wo, Hc, 1, bo, hG, Hc, qb, Hc, (int)BL, Hc, Hc, 1.0f, false);
      // mid = gelu(h2 @ W1 + c1)
      gemm(qb, Hc, W1, 2 * Hc, 1, c1, nullptr, 0, mid, 2 * Hc, (int)BL, 2 * Hc, Hc, 1.0f, true);
      // eo = h2 + (mid @ W2 + c2)
      gemm(mid, 2 * Hc, W2, Hc, 1, c2, qb, Hc, eo, Hc, (int)BL, Hc, 2 * Hc, 1.0f, false);
      axpy_expert<<<blocks(BL * Hc), 256, 0, stream>>>(acc, eo, wbuf, 4 + j, (long long)L * Hc, BL * Hc);
    }

    // y (B,H,L) = x + acc^T
    final_out<<<blocks(BL * Hc), 256, 0, stream>>>(x, acc, ys[lvl], L, BL * Hc);
  }

  usage_kernel<<<1, 32, 0, stream>>>(counts, usageOut);
}
